// MambaLM_86912958202094
// MI455X (gfx1250) — matrix operand order repaired and checked by reading, not compiled, not run on hardware
//
#include <hip/hip_runtime.h>
#include <math.h>

// ---------------- model dims ----------------
#define DMODEL 1024
#define NLAYER 4
#define DSTATE 16
#define BATCH  2
#define SEQL   1024
#define DINNER 2048            // EXP * DMODEL
#define DTRANK 64
#define DCONV  4
#define MTOK   (BATCH * SEQL)  // 2048 rows for every GEMM
#define VOCAB  32000
#define LNEPS  1e-5f

// ---------------- CDNA5 async-copy availability ----------------
#if defined(__gfx1250__) && \
    __has_builtin(__builtin_amdgcn_global_load_async_to_lds_b128) && \
    __has_builtin(__builtin_amdgcn_s_wait_asynccnt)
#define USE_ASYNC 1
#else
#define USE_ASYNC 0
#endif

// exact pointee type of the async builtin's pointer params (per clang diagnostic):
// '__attribute__((__vector_size__(4*sizeof(int)))) int' in AS(1) / AS(3)
typedef int v4i_vs __attribute__((vector_size(16)));

// ---------------- WMMA types ----------------
typedef __attribute__((ext_vector_type(16))) _Float16 v16h;
typedef __attribute__((ext_vector_type(8)))  _Float16 v8h;
typedef __attribute__((ext_vector_type(8)))  float    v8f;

union HFrag { v16h v; v8h h[2]; };

// ---------------- GEMM tiling ----------------
#define BM 128
#define BN 128
#define BK 32
#define BKP 40   // padded LDS row (f16 elems): 80B rows keep 16B alignment, no bank conflicts

#define EPI_NONE     0
#define EPI_CLIP     1
#define EPI_SOFTPLUS 2
#define EPI_RESID    3

// Stage one 128-row x 32-col f16 tile (row-major, leading dim `ld` elements)
// into padded LDS. Async path: global->LDS DMA, 2 instructions/wave, each
// covering 8 rows x 4 x 16B chunks. Fallback: grouped b128 loads then stores.
__device__ __forceinline__ void stage_tile(const _Float16* __restrict__ gsrc,
                                           int ld, _Float16* ldst,
                                           int rowsValid, int tid)
{
#if USE_ASYNC
    const int lane = tid & 31;
    const int wv   = tid >> 5;
    #pragma unroll
    for (int i = 0; i < 2; ++i) {
        const int row = wv * 16 + i * 8 + (lane >> 2);
        const int ch  = lane & 3;
        if (row < rowsValid) {
            const _Float16* g = gsrc + (size_t)row * ld + ch * 8;
            _Float16*       l = ldst + row * BKP + ch * 8;
            __builtin_amdgcn_global_load_async_to_lds_b128(
                (__attribute__((address_space(1))) v4i_vs*)(uintptr_t)g,
                (__attribute__((address_space(3))) v4i_vs*)(uint32_t)(uintptr_t)l,
                0, 0);
        }
    }
#else
    uint4 v[2]; int row[2], ch[2];
    #pragma unroll
    for (int i = 0; i < 2; ++i) {            // issue both loads first (MLP)
        int c = tid + i * 256;               // 0..511 chunk id
        row[i] = c >> 2; ch[i] = c & 3;
        v[i] = make_uint4(0u, 0u, 0u, 0u);
        if (row[i] < rowsValid)
            v[i] = *(const uint4*)(gsrc + (size_t)row[i] * ld + ch[i] * 8);
    }
    #pragma unroll
    for (int i = 0; i < 2; ++i)
        *(uint4*)(ldst + row[i] * BKP + ch[i] * 8) = v[i];
#endif
}

// C(MxN) = epilogue(A(MxK,f16,lda) * Wt(NxK,f16,ldw)^T + bias [+resid])
// M % 128 == 0, K % 32 == 0; N arbitrary multiple of 4 (tail tile guarded).
__global__ __launch_bounds__(256)
void gemm_f16wmma(const _Float16* __restrict__ A, int lda,
                  const _Float16* __restrict__ Wt, int ldw,
                  int N, int K,
                  const float* __restrict__ bias,
                  const float* __restrict__ resid,
                  float* __restrict__ C, _Float16* __restrict__ Ch,
                  int ldc, int mode)
{
    __shared__ __align__(16) _Float16 AsS[2][BM][BKP];
    __shared__ __align__(16) _Float16 BsS[2][BN][BKP];

    const int tid  = threadIdx.x;
    const int lane = tid & 31;
    const int wid  = tid >> 5;      // 8 wave32s
    const int wm   = wid & 3;       // 4 waves along M (4*32 = 128)
    const int wn   = wid >> 2;      // 2 waves along N (2*64 = 128)
    const int hh   = lane >> 4;     // half-wave select
    const int rc   = lane & 15;     // row (A) / col (B,C) inside 16x16 tile

    const int m0 = blockIdx.y * BM;
    const int n0 = blockIdx.x * BN;
    const int nvalid = (N - n0 < BN) ? (N - n0) : BN;

#if USE_ASYNC
    // async path never writes rows >= nvalid: zero them once (both buffers)
    if (nvalid < BN) {
        for (int r = nvalid + tid; r < BN; r += 256)
            #pragma unroll
            for (int c = 0; c < BKP; ++c) {
                BsS[0][r][c] = (_Float16)0;
                BsS[1][r][c] = (_Float16)0;
            }
    }
#endif

    const _Float16* Abase = A  + (size_t)m0 * lda;
    const _Float16* Bbase = Wt + (size_t)n0 * ldw;

    v8f acc[2][4] = {};

    // prologue: stage tile 0 into buffer 0
    stage_tile(Abase, lda, &AsS[0][0][0], BM, tid);
    stage_tile(Bbase, ldw, &BsS[0][0][0], nvalid, tid);
#if USE_ASYNC
    __builtin_amdgcn_s_wait_asynccnt(0);
#endif
    __syncthreads();

    const int kTiles = K / BK;
    for (int kt = 0; kt < kTiles; ++kt) {
        const int cur = kt & 1;
        if (kt + 1 < kTiles) {  // overlap next tile's global->LDS with WMMA
            stage_tile(Abase + (kt + 1) * BK, lda, &AsS[cur ^ 1][0][0], BM, tid);
            stage_tile(Bbase + (kt + 1) * BK, ldw, &BsS[cur ^ 1][0][0], nvalid, tid);
        }

        // A 16x32 f16 fragment (ISA layout): lanes 0-15 hold K 0..7 & 16..23,
        // lanes 16-31 hold K 8..15 & 24..31 -> v8h chunks at 8*hh and 16+8*hh
        HFrag af[2];
        #pragma unroll
        for (int i = 0; i < 2; ++i) {
            int row = wm * 32 + i * 16 + rc;
            af[i].h[0] = *(const v8h*)&AsS[cur][row][8 * hh];
            af[i].h[1] = *(const v8h*)&AsS[cur][row][16 + 8 * hh];
        }
        // B 32x16 fragment: lanes 0-15 hold K 0..7 & 16..23, lanes 16-31 hold K 8..15 & 24..31
        HFrag bf[4];
        #pragma unroll
        for (int j = 0; j < 4; ++j) {
            int col = wn * 64 + j * 16 + rc;
            bf[j].h[0] = *(const v8h*)&BsS[cur][col][8 * hh];
            bf[j].h[1] = *(const v8h*)&BsS[cur][col][16 + 8 * hh];
        }
        #pragma unroll
        for (int i = 0; i < 2; ++i)
            #pragma unroll
            for (int j = 0; j < 4; ++j)
                acc[i][j] = __builtin_amdgcn_wmma_f32_16x16x32_f16(
                    false, af[i].v, false, bf[j].v, (short)0, acc[i][j],
                    false, false);

#if USE_ASYNC
        __builtin_amdgcn_s_wait_asynccnt(0);
#endif
        __syncthreads();
    }

    // epilogue: C/D layout: lanes0-15 -> M=r, N=lane; lanes16-31 -> M=r+8, N=lane-16
    #pragma unroll
    for (int i = 0; i < 2; ++i) {
        #pragma unroll
        for (int j = 0; j < 4; ++j) {
            int gm0 = m0 + wm * 32 + i * 16 + 8 * hh;
            int gn  = n0 + wn * 64 + j * 16 + rc;
            if (gn >= N) continue;
            float bv = bias ? bias[gn] : 0.0f;
            #pragma unroll
            for (int r = 0; r < 8; ++r) {
                int gm = gm0 + r;
                float v = acc[i][j][r] + bv;
                if (mode == EPI_CLIP)          v = fminf(1.0f, fmaxf(-1.0f, v));
                else if (mode == EPI_SOFTPLUS) v = (v > 20.0f) ? v : log1pf(expf(v));
                else if (mode == EPI_RESID)    v += resid[(size_t)gm * ldc + gn];
                if (C)  C[(size_t)gm * ldc + gn]  = v;
                if (Ch) Ch[(size_t)gm * ldc + gn] = (_Float16)v;
            }
        }
    }
}

// ---------------- weight transpose + fp32 -> f16 (K x N -> N x K) ----------
// K, N multiples of 32. One-shot per call; amortized across 16 M-tiles.
__global__ __launch_bounds__(256)
void transpose_convert(const float* __restrict__ in, _Float16* __restrict__ out,
                       int K, int N)
{
    __shared__ float tile[32][33];
    const int kb = blockIdx.y * 32, nb = blockIdx.x * 32;
    const int tx = threadIdx.x & 31, ty = threadIdx.x >> 5;   // 32 x 8
    #pragma unroll
    for (int i = 0; i < 32; i += 8)
        tile[ty + i][tx] = in[(size_t)(kb + ty + i) * N + nb + tx];
    __syncthreads();
    #pragma unroll
    for (int i = 0; i < 32; i += 8)
        out[(size_t)(nb + ty + i) * K + kb + tx] = (_Float16)tile[tx][ty + i];
}

// ---------------- embedding gather ----------------
__global__ __launch_bounds__(256)
void embed_kernel(const int* __restrict__ tokens, const float* __restrict__ emb,
                  float* __restrict__ x)
{
    int row = blockIdx.x;
    int tok = tokens[row];
    const float* e = emb + (size_t)tok * DMODEL;
    float* o = x + (size_t)row * DMODEL;
    for (int i = threadIdx.x; i < DMODEL; i += 256) o[i] = e[i];
}

// ---------------- layernorm (one block per row) -> f16 GEMM operand --------
__global__ __launch_bounds__(256)
void ln_kernel(const float* __restrict__ x, const float* __restrict__ w,
               const float* __restrict__ b, _Float16* __restrict__ out)
{
    __shared__ float red[256];
    const int row = blockIdx.x;
    const float* xr = x + (size_t)row * DMODEL;

    float s = 0.0f;
    for (int i = threadIdx.x; i < DMODEL; i += 256) s += xr[i];
    red[threadIdx.x] = s;
    __syncthreads();
    for (int st = 128; st > 0; st >>= 1) {
        if (threadIdx.x < st) red[threadIdx.x] += red[threadIdx.x + st];
        __syncthreads();
    }
    const float mean = red[0] * (1.0f / DMODEL);
    __syncthreads();

    float v = 0.0f;
    for (int i = threadIdx.x; i < DMODEL; i += 256) {
        float d = xr[i] - mean;
        v += d * d;
    }
    red[threadIdx.x] = v;
    __syncthreads();
    for (int st = 128; st > 0; st >>= 1) {
        if (threadIdx.x < st) red[threadIdx.x] += red[threadIdx.x + st];
        __syncthreads();
    }
    const float inv = rsqrtf(red[0] * (1.0f / DMODEL) + LNEPS);

    _Float16* o = out + (size_t)row * DMODEL;
    for (int i = threadIdx.x; i < DMODEL; i += 256)
        o[i] = (_Float16)((xr[i] - mean) * inv * w[i] + b[i]);
}

// ---------------- causal depthwise conv (k=4) + SiLU ----------------
__global__ __launch_bounds__(256)
void conv_silu_kernel(const float* __restrict__ xz, const float* __restrict__ cw,
                      const float* __restrict__ cb, float* __restrict__ out,
                      _Float16* __restrict__ outh)
{
    int idx = blockIdx.x * 256 + threadIdx.x;     // over BATCH*SEQL*DINNER
    int c   = idx % DINNER;
    int row = idx / DINNER;                       // b*SEQL + l
    int l   = row % SEQL;
    float acc = cb[c];
    #pragma unroll
    for (int k = 0; k < DCONV; ++k) {
        int d = (DCONV - 1) - k;                  // how far back
        if (l - d >= 0)
            acc += cw[c * DCONV + k] * xz[(size_t)(row - d) * (2 * DINNER) + c];
    }
    float sv = acc / (1.0f + __expf(-acc));       // silu
    out[idx]  = sv;
    outh[idx] = (_Float16)sv;
}

// ---------------- selective scan (seq over L, parallel over B*DINNER) ------
__global__ __launch_bounds__(256)
void scan_kernel(const float* __restrict__ dt,   // (MTOK, DINNER)
                 const float* __restrict__ xz,   // (MTOK, 2*DINNER): z = 2nd half
                 const float* __restrict__ xcs,  // (MTOK, DINNER)
                 const float* __restrict__ dbl,  // (MTOK, 96): B @64, C @80
                 const float* __restrict__ A_log,// (DINNER, DSTATE)
                 const float* __restrict__ Dp,   // (DINNER)
                 _Float16* __restrict__ y)       // (MTOK, DINNER) f16 GEMM operand
{
    int gid = blockIdx.x * blockDim.x + threadIdx.x;  // 0..BATCH*DINNER-1
    int d = gid % DINNER;
    int b = gid / DINNER;

    float Av[DSTATE], h[DSTATE];
    #pragma unroll
    for (int s = 0; s < DSTATE; ++s) {
        Av[s] = -__expf(A_log[d * DSTATE + s]);
        h[s]  = 0.0f;
    }
    const float Dd = Dp[d];

    for (int l = 0; l < SEQL; ++l) {
        size_t row = (size_t)b * SEQL + l;
        float dtv = dt[row * DINNER + d];
        float u   = xcs[row * DINNER + d];
        float zv  = xz[row * (2 * DINNER) + DINNER + d];
        const float* bc = dbl + row * 96;
        float yv = 0.0f;
        #pragma unroll
        for (int s = 0; s < DSTATE; ++s) {
            float dA = __expf(dtv * Av[s]);
            h[s] = h[s] * dA + (dtv * bc[64 + s]) * u;
            yv += h[s] * bc[80 + s];
        }
        yv += u * Dd;
        float sig = 1.0f / (1.0f + __expf(-zv));
        y[row * DINNER + d] = (_Float16)(yv * (zv * sig));
    }
}

// ---------------- host orchestration ----------------
extern "C" void kernel_launch(void* const* d_in, const int* in_sizes, int n_in,
                              void* d_out, int out_size, void* d_ws, size_t ws_size,
                              hipStream_t stream)
{
    (void)in_sizes; (void)n_in; (void)out_size; (void)ws_size;

    const int*   tokens = (const int*)d_in[0];
    const float* emb    = (const float*)d_in[1];
    const float* ln1w   = (const float*)d_in[2];
    const float* ln1b   = (const float*)d_in[3];
    const float* ln2w   = (const float*)d_in[4];
    const float* ln2b   = (const float*)d_in[5];
    const float* lin1w  = (const float*)d_in[6];
    const float* lin1b  = (const float*)d_in[7];
    const float* lin2w  = (const float*)d_in[8];
    const float* lin2b  = (const float*)d_in[9];
    const float* ipw    = (const float*)d_in[10];
    const float* convw  = (const float*)d_in[11];
    const float* convb  = (const float*)d_in[12];
    const float* xpw    = (const float*)d_in[13];
    const float* dtw    = (const float*)d_in[14];
    const float* dtbias = (const float*)d_in[15];
    const float* A_log  = (const float*)d_in[16];
    const float* Dparam = (const float*)d_in[17];
    const float* opw    = (const float*)d_in[18];
    const float* lnow   = (const float*)d_in[19];
    const float* lnob   = (const float*)d_in[20];
    const float* projw  = (const float*)d_in[21];
    const float* projb  = (const float*)d_in[22];

    // ---- fp32 activation scratch ----
    float* wsf = (float*)d_ws;
    float* x    = wsf; wsf += (size_t)MTOK * DMODEL;   // residual stream
    float* xz   = wsf; wsf += (size_t)MTOK * 4096;     // in_proj out (xc|z)
    float* xcs  = wsf; wsf += (size_t)MTOK * DINNER;   // conv+silu (f32 for scan)
    float* dblb = wsf; wsf += (size_t)MTOK * 96;       // x_proj out (dt|B|C)
    float* dtb  = wsf; wsf += (size_t)MTOK * DINNER;   // softplus dt

    // ---- f16 arena: transposed weights + GEMM A operands ----
    _Float16* h = (_Float16*)wsf;
    _Float16* lin1t = h; h += (size_t)NLAYER * 2048 * 1024;   // N x K
    _Float16* lin2t = h; h += (size_t)NLAYER * 1024 * 2048;
    _Float16* ipwt  = h; h += (size_t)NLAYER * 4096 * 1024;
    _Float16* xpwt  = h; h += (size_t)NLAYER * 96   * 2048;
    _Float16* dtwt  = h; h += (size_t)NLAYER * 2048 * 64;
    _Float16* opwt  = h; h += (size_t)NLAYER * 1024 * 2048;
    _Float16* projt = h; h += (size_t)VOCAB  * 1024;
    _Float16* t0h   = h; h += (size_t)MTOK * DMODEL;          // LN out
    _Float16* gh    = h; h += (size_t)MTOK * 2048;            // MLP hidden
    _Float16* xcsh  = h; h += (size_t)MTOK * DINNER;
    _Float16* dblh  = h; h += (size_t)MTOK * 96;
    _Float16* ybh   = h; h += (size_t)MTOK * DINNER;

    const dim3 blk(256);
    const int mGrid = MTOK / BM;   // 16

    // ---- one-shot weight transpose/convert (K x N fp32 -> N x K f16) ----
    for (int i = 0; i < NLAYER; ++i) {
        transpose_convert<<<dim3(2048 / 32, 1024 / 32), blk, 0, stream>>>(
            lin1w + (size_t)i * 1024 * 2048, lin1t + (size_t)i * 2048 * 1024, 1024, 2048);
        transpose_convert<<<dim3(1024 / 32, 2048 / 32), blk, 0, stream>>>(
            lin2w + (size_t)i * 2048 * 1024, lin2t + (size_t)i * 1024 * 2048, 2048, 1024);
        transpose_convert<<<dim3(4096 / 32, 1024 / 32), blk, 0, stream>>>(
            ipw + (size_t)i * 1024 * 4096, ipwt + (size_t)i * 4096 * 1024, 1024, 4096);
        transpose_convert<<<dim3(96 / 32, 2048 / 32), blk, 0, stream>>>(
            xpw + (size_t)i * 2048 * 96, xpwt + (size_t)i * 96 * 2048, 2048, 96);
        transpose_convert<<<dim3(2048 / 32, 64 / 32), blk, 0, stream>>>(
            dtw + (size_t)i * 64 * 2048, dtwt + (size_t)i * 2048 * 64, 64, 2048);
        transpose_convert<<<dim3(1024 / 32, 2048 / 32), blk, 0, stream>>>(
            opw + (size_t)i * 2048 * 1024, opwt + (size_t)i * 1024 * 2048, 2048, 1024);
    }
    transpose_convert<<<dim3(VOCAB / 32, 1024 / 32), blk, 0, stream>>>(
        projw, projt, 1024, VOCAB);

    embed_kernel<<<MTOK, blk, 0, stream>>>(tokens, emb, x);

    for (int i = 0; i < NLAYER; ++i) {
        // gated MLP branch
        ln_kernel<<<MTOK, blk, 0, stream>>>(x, ln1w + i * DMODEL, ln1b + i * DMODEL, t0h);
        gemm_f16wmma<<<dim3(2048 / BN, mGrid), blk, 0, stream>>>(
            t0h, DMODEL, lin1t + (size_t)i * 2048 * 1024, 1024, 2048, 1024,
            lin1b + (size_t)i * 2048, nullptr, nullptr, gh, 2048, EPI_CLIP);
        gemm_f16wmma<<<dim3(DMODEL / BN, mGrid), blk, 0, stream>>>(
            gh, 2048, lin2t + (size_t)i * 1024 * 2048, 2048, 1024, 2048,
            lin2b + (size_t)i * DMODEL, x, x, nullptr, 1024, EPI_RESID);

        // Mamba branch
        ln_kernel<<<MTOK, blk, 0, stream>>>(x, ln2w + i * DMODEL, ln2b + i * DMODEL, t0h);
        gemm_f16wmma<<<dim3(4096 / BN, mGrid), blk, 0, stream>>>(
            t0h, DMODEL, ipwt + (size_t)i * 4096 * 1024, 1024, 4096, 1024,
            nullptr, nullptr, xz, nullptr, 4096, EPI_NONE);
        conv_silu_kernel<<<(MTOK * DINNER) / 256, blk, 0, stream>>>(
            xz, convw + (size_t)i * DINNER * DCONV, convb + (size_t)i * DINNER,
            xcs, xcsh);
        gemm_f16wmma<<<dim3(1, mGrid), blk, 0, stream>>>(
            xcsh, DINNER, xpwt + (size_t)i * 96 * 2048, 2048, 96, 2048,
            nullptr, nullptr, dblb, dblh, 96, EPI_NONE);
        gemm_f16wmma<<<dim3(DINNER / BN, mGrid), blk, 0, stream>>>(
            dblh, 96, dtwt + (size_t)i * 2048 * 64, 64, 2048, 64,
            dtbias + (size_t)i * DINNER, nullptr, dtb, nullptr, 2048, EPI_SOFTPLUS);
        scan_kernel<<<(BATCH * DINNER) / 256, blk, 0, stream>>>(
            dtb, xz, xcs, dblb, A_log + (size_t)i * DINNER * DSTATE,
            Dparam + (size_t)i * DINNER, ybh);
        gemm_f16wmma<<<dim3(DMODEL / BN, mGrid), blk, 0, stream>>>(
            ybh, 2048, opwt + (size_t)i * 1024 * 2048, 2048, 1024, 2048,
            nullptr, x, x, nullptr, 1024, EPI_RESID);
    }

    // final LN + vocab projection (dominant GEMM: 2048 x 32000 x 1024)
    ln_kernel<<<MTOK, blk, 0, stream>>>(x, lnow, lnob, t0h);
    gemm_f16wmma<<<dim3(VOCAB / BN, mGrid), blk, 0, stream>>>(
        t0h, DMODEL, projt, 1024, VOCAB, 1024, projb, nullptr,
        (float*)d_out, nullptr, VOCAB, EPI_NONE);
}
